// Attention_60370060313311
// MI455X (gfx1250) — matrix operand order repaired and checked by reading, not compiled, not run on hardware
//
#include <hip/hip_runtime.h>

// ---------------------------------------------------------------------------
// Sliding-window attention for MI455X (gfx1250), bf16 WMMA pipeline.
//   B=2, T=2048, C=1024, H=16, HD=64, WINDOW=524
// v2: double-buffered GEMM k-loops (overlap loads with WMMA), V stored
//     transposed (b,h,hd,t) for contiguous attention fragments, explicit
//     vector loads, global_prefetch of streaming K/V.
// ---------------------------------------------------------------------------

typedef __attribute__((ext_vector_type(16))) __bf16 v16bf;
typedef __attribute__((ext_vector_type(8)))  __bf16 v8bf;
typedef __attribute__((ext_vector_type(8)))  float  v8f;

#define BATCH 2
#define SEQ   2048
#define CH    1024
#define NH    16
#define HD    64
#define WIN   524          // 16 + ((6/12)^2)*(2048-16)
#define MROWS (BATCH * SEQ)     // 4096
#define KT32  (CH / 32)         // 32 k-tiles of 32

static __device__ __forceinline__ v8f wmma_bf16(v16bf a, v16bf b, v8f c) {
    return __builtin_amdgcn_wmma_f32_16x16x32_bf16(false, a, false, b, (short)0, c,
                                                   false, false);
}

static __device__ __forceinline__ v16bf cat8(v8bf lo, v8bf hi) {
    return __builtin_shufflevector(lo, hi, 0, 1, 2, 3, 4, 5, 6, 7,
                                   8, 9, 10, 11, 12, 13, 14, 15);
}

// A-fragment k map (ISA 7.12.2, 16-bit A 16x32): lane base = half*8,
// j<8 -> base+j ; j>=8 -> base+16+(j-8).  Element runs are 8-contiguous.
// ---------------------------------------------------------------------------
// Pack f32 (M x K, row-major) into bf16 A-fragments: [mt][kt][lane][16]
// ---------------------------------------------------------------------------
__global__ void __launch_bounds__(128) pack_a_kernel(const float* __restrict__ src,
                                                     __bf16* __restrict__ dst,
                                                     int M, int K) {
    int w = (blockIdx.x * blockDim.x + threadIdx.x) >> 5;
    int lane = threadIdx.x & 31;
    int ktiles = K >> 5;
    int nfrag = (M >> 4) * ktiles;
    if (w >= nfrag) return;
    int mt = w / ktiles, kt = w - mt * ktiles;
    int row = mt * 16 + (lane & 15);
    int kb  = kt * 32 + (lane >> 4) * 8;
    const float* s = src + (size_t)row * K + kb;
    v16bf fr;
#pragma unroll
    for (int j = 0; j < 8; ++j) fr[j]     = (__bf16)s[j];
#pragma unroll
    for (int j = 0; j < 8; ++j) fr[8 + j] = (__bf16)s[16 + j];
    *(v16bf*)(dst + (size_t)w * 512 + lane * 16) = fr;
}

// ---------------------------------------------------------------------------
// Pack f32 W (K x N, row-major) into bf16 B-fragments: [nt][kt][lane][16]
// B layout (ISA): lane l holds n = l&15, k = (l>>4)*8 + j (j<8), (l>>4)*8 + 16 + (j-8) (j>=8)
// ---------------------------------------------------------------------------
__global__ void __launch_bounds__(128) pack_b_kernel(const float* __restrict__ src,
                                                     __bf16* __restrict__ dst,
                                                     int K, int N) {
    int w = (blockIdx.x * blockDim.x + threadIdx.x) >> 5;
    int lane = threadIdx.x & 31;
    int ktiles = K >> 5;
    int nfrag = (N >> 4) * ktiles;
    if (w >= nfrag) return;
    int nt = w / ktiles, kt = w - nt * ktiles;
    int n  = nt * 16 + (lane & 15);
    int k0 = kt * 32 + (lane >> 4) * 8;
    v16bf fr;
#pragma unroll
    for (int j = 0; j < 16; ++j) fr[j] = (__bf16)src[(size_t)(k0 + (j < 8 ? j : j + 8)) * N + n];
    *(v16bf*)(dst + (size_t)w * 512 + lane * 16) = fr;
}

// ---------------------------------------------------------------------------
// Fused QKV projection. One wave -> 16 rows x 64 cols (one head) of q,k,v.
// Double-buffered k-loop per n-tile so loads overlap WMMA.
// RoPE applied in-register to q,k. q,k stored (b,h,t,hd); v stored transposed
// (b,h,hd,t) via an LDS staging tile for contiguous attention loads.
// ---------------------------------------------------------------------------
__global__ void __launch_bounds__(256) qkv_kernel(const __bf16* __restrict__ pX,
                                                  const __bf16* __restrict__ pWq,
                                                  const __bf16* __restrict__ pWk,
                                                  const __bf16* __restrict__ pWv,
                                                  const float*  __restrict__ rope_cos,
                                                  const float*  __restrict__ rope_sin,
                                                  __bf16* __restrict__ qout,
                                                  __bf16* __restrict__ kout,
                                                  __bf16* __restrict__ vT) {
    __shared__ __bf16 vstage[8][16 * 64];
    int w = (blockIdx.x * 256 + threadIdx.x) >> 5;   // 4096 waves total
    int wib  = (threadIdx.x >> 5);
    int lane = threadIdx.x & 31;
    int mt = w >> 4;          // 0..255
    int nc = w & 15;          // head index, 64-wide n chunk

    v8f aq[4], ak[4], av[4];
    const __bf16* Abase = pX + (size_t)mt * KT32 * 512 + lane * 16;

#pragma unroll
    for (int jn = 0; jn < 4; ++jn) {
        size_t boff = (size_t)(nc * 4 + jn) * KT32 * 512 + lane * 16;
        const __bf16* Bq = pWq + boff;
        const __bf16* Bk = pWk + boff;
        const __bf16* Bv = pWv + boff;
        v8f cq = {}, ck = {}, cv = {};
        v16bf a0 = *(const v16bf*)(Abase);
        v16bf q0 = *(const v16bf*)(Bq);
        v16bf k0 = *(const v16bf*)(Bk);
        v16bf v0 = *(const v16bf*)(Bv);
        for (int kt = 0; kt < KT32 - 1; ++kt) {
            // issue next k-step loads before this step's WMMAs
            v16bf a1 = *(const v16bf*)(Abase + (size_t)(kt + 1) * 512);
            v16bf q1 = *(const v16bf*)(Bq + (size_t)(kt + 1) * 512);
            v16bf k1 = *(const v16bf*)(Bk + (size_t)(kt + 1) * 512);
            v16bf v1 = *(const v16bf*)(Bv + (size_t)(kt + 1) * 512);
            cq = wmma_bf16(a0, q0, cq);
            ck = wmma_bf16(a0, k0, ck);
            cv = wmma_bf16(a0, v0, cv);
            a0 = a1; q0 = q1; k0 = k1; v0 = v1;
        }
        cq = wmma_bf16(a0, q0, cq);
        ck = wmma_bf16(a0, k0, ck);
        cv = wmma_bf16(a0, v0, cv);
        aq[jn] = cq; ak[jn] = ck; av[jn] = cv;
    }

    int half = lane >> 4, col = lane & 15;
    int m0 = mt * 16;
    int b  = m0 >> 11;                 // /SEQ
    int tbase = m0 & (SEQ - 1);

    // RoPE: pair (hd, hd+32) -> acc tiles jn and jn+2, same lane/slot.
    // out1 = x1*c + x2*s ; out2 = x1*c - x2*s   (matches reference)
#pragma unroll
    for (int i = 0; i < 8; ++i) {
        int t = tbase + i + half * 8;
#pragma unroll
        for (int jn = 0; jn < 2; ++jn) {
            int hh = jn * 16 + col;                 // 0..31
            float c = rope_cos[t * (HD / 2) + hh];
            float s = rope_sin[t * (HD / 2) + hh];
            float x1 = aq[jn][i], x2 = aq[jn + 2][i];
            aq[jn][i]     = x1 * c + x2 * s;
            aq[jn + 2][i] = x1 * c - x2 * s;
            x1 = ak[jn][i]; x2 = ak[jn + 2][i];
            ak[jn][i]     = x1 * c + x2 * s;
            ak[jn + 2][i] = x1 * c - x2 * s;
        }
    }

    // q,k -> (b,h,t,hd) (coalesced 32B per half-wave); v -> LDS stage
    __bf16* vs = vstage[wib];
#pragma unroll
    for (int i = 0; i < 8; ++i) {
        int r = i + half * 8;
        int t = tbase + r;
        size_t base = (((size_t)b * NH + nc) * SEQ + t) * HD;
#pragma unroll
        for (int jn = 0; jn < 4; ++jn) {
            int hd = jn * 16 + col;
            qout[base + hd] = (__bf16)aq[jn][i];
            kout[base + hd] = (__bf16)ak[jn][i];
            vs[r * 64 + hd] = (__bf16)av[jn][i];
        }
    }
    __builtin_amdgcn_wave_barrier();

    // v transposed: vT[((b*NH+h)*HD + hd)*SEQ + t], 32B contiguous per lane
#pragma unroll
    for (int r = 0; r < 2; ++r) {
        int hd = lane + r * 32;
        v16bf fr;
#pragma unroll
        for (int j = 0; j < 16; ++j) fr[j] = vs[j * 64 + hd];
        *(v16bf*)(vT + (((size_t)b * NH + nc) * HD + hd) * SEQ + tbase) = fr;
    }
}

// ---------------------------------------------------------------------------
// Flash-style sliding-window attention. One wave per (b, h, 16-row q tile).
// Writes normalized output directly as bf16 A-fragments for the final GEMM.
// ---------------------------------------------------------------------------
__global__ void __launch_bounds__(128) attn_kernel(const __bf16* __restrict__ q,
                                                   const __bf16* __restrict__ k,
                                                   const __bf16* __restrict__ vT,
                                                   __bf16* __restrict__ packedY) {
    __shared__ __bf16 lds[4][16 * 64];
    int wib  = threadIdx.x >> 5;
    int lane = threadIdx.x & 31;
    int w = blockIdx.x * 4 + wib;                 // 0..4095
    int b   = w >> 11;                            // /(NH*128)
    int rem = w & 2047;
    int h   = rem >> 7;
    int qt  = rem & 127;
    int t0  = qt * 16;
    int half = lane >> 4, col = lane & 15;

    const __bf16* Qh = q  + (((size_t)b * NH + h) * SEQ) * HD;
    const __bf16* Kh = k  + (((size_t)b * NH + h) * SEQ) * HD;
    const __bf16* Vt = vT + (((size_t)b * NH + h) * HD) * SEQ;

    // Q A-fragments (rows t0..t0+15, hd split into two k-chunks of 32)
    v16bf aq[2];
#pragma unroll
    for (int kc = 0; kc < 2; ++kc) {
        const __bf16* src = Qh + (size_t)(t0 + col) * HD + kc * 32 + half * 8;
        aq[kc] = cat8(*(const v8bf*)(src), *(const v8bf*)(src + 16));
    }

    v8f O[4] = {v8f{}, v8f{}, v8f{}, v8f{}};
    float mrow[8], lrow[8];
#pragma unroll
    for (int i = 0; i < 8; ++i) { mrow[i] = -1e30f; lrow[i] = 0.0f; }

    __bf16* P = lds[wib];                         // 16 x 32 staging (16x64 at end)

    int s_lo = t0 - (WIN - 1); if (s_lo < 0) s_lo = 0;
    int s0 = s_lo & ~31;
    int s_end = t0 + 15;
    // note: s0 is 32-aligned and <= t0+15 <= 2047, so all keys in [s0, s0+31]
    // stay within [0, SEQ-1]; no clamping needed.

    for (; s0 <= s_end; s0 += 32) {
        // prefetch next key block's K and V streams (global_prefetch_b8)
        __builtin_prefetch(Kh + (size_t)(s0 + 32 + col) * HD, 0, 0);
        __builtin_prefetch(Vt + (size_t)(half * 32 + col) * SEQ + s0 + 32, 0, 0);
#pragma unroll
        for (int sub = 0; sub < 2; ++sub) {
            int ks = s0 + sub * 16;
            v8f sacc = {};
#pragma unroll
            for (int kc = 0; kc < 2; ++kc) {
                const __bf16* src = Kh + (size_t)(ks + col) * HD + kc * 32 + half * 8;
                v16bf bk = cat8(*(const v8bf*)(src), *(const v8bf*)(src + 16));
                sacc = wmma_bf16(aq[kc], bk, sacc);
            }
#pragma unroll
            for (int i = 0; i < 8; ++i) {
                int t   = t0 + i + half * 8;
                int key = ks + col;
                bool ok = (key <= t) && (key > t - WIN);
                float sv = ok ? sacc[i] * 0.125f : -1e30f;
                float mx = sv;
#pragma unroll
                for (int d = 1; d < 16; d <<= 1)
                    mx = fmaxf(mx, __shfl_xor(mx, d, 32));
                float mnew  = fmaxf(mrow[i], mx);
                float alpha = __expf(mrow[i] - mnew);
                float p = ok ? __expf(sv - mnew) : 0.0f;
                float ps = p;
#pragma unroll
                for (int d = 1; d < 16; d <<= 1)
                    ps += __shfl_xor(ps, d, 32);
                lrow[i] = lrow[i] * alpha + ps;
                mrow[i] = mnew;
                O[0][i] *= alpha; O[1][i] *= alpha;
                O[2][i] *= alpha; O[3][i] *= alpha;
                P[(i + half * 8) * 32 + sub * 16 + col] = (__bf16)p;
            }
        }
        __builtin_amdgcn_wave_barrier();          // keep LDS write->read ordering
        // P (16x32) as A-fragment, then P @ V over 32 keys
        const __bf16* pr = P + col * 32 + half * 8;
        v16bf ap = cat8(*(const v8bf*)(pr), *(const v8bf*)(pr + 16));
#pragma unroll
        for (int jn = 0; jn < 4; ++jn) {
            // V^T: two runs of 8 keys per lane (table order) at fixed hd = jn*16+col
            const __bf16* src = Vt + (size_t)(jn * 16 + col) * SEQ + s0 + half * 8;
            v16bf bv = cat8(*(const v8bf*)(src), *(const v8bf*)(src + 16));
            O[jn] = wmma_bf16(ap, bv, O[jn]);
        }
        __builtin_amdgcn_wave_barrier();
    }

    // Normalize and stage output tile in LDS (16 x 64 bf16)
#pragma unroll
    for (int i = 0; i < 8; ++i) {
        float inv = 1.0f / lrow[i];
        O[0][i] *= inv; O[1][i] *= inv; O[2][i] *= inv; O[3][i] *= inv;
    }
    __bf16* Ob = lds[wib];
#pragma unroll
    for (int i = 0; i < 8; ++i)
#pragma unroll
        for (int jn = 0; jn < 4; ++jn)
            Ob[(i + half * 8) * 64 + jn * 16 + col] = (__bf16)O[jn][i];
    __builtin_amdgcn_wave_barrier();

    // Emit A-fragments of y for the output projection: [mt][kt][lane][16]
    int mt = (b << 7) + qt;                       // b*(SEQ/16) + qt
#pragma unroll
    for (int kc = 0; kc < 2; ++kc) {
        const __bf16* orow = Ob + col * 64 + kc * 32 + half * 8;
        v16bf fr = cat8(*(const v8bf*)(orow), *(const v8bf*)(orow + 16));
        *(v16bf*)(packedY + ((size_t)mt * KT32 + (h * 2 + kc)) * 512 + lane * 16) = fr;
    }
}

// ---------------------------------------------------------------------------
// Output projection: packedY (bf16 A-frags) x packedWo -> f32 (B*T, C)
// Double-buffered k-loop.
// ---------------------------------------------------------------------------
__global__ void __launch_bounds__(256) outproj_kernel(const __bf16* __restrict__ pY,
                                                      const __bf16* __restrict__ pWo,
                                                      float* __restrict__ out) {
    int w = (blockIdx.x * 256 + threadIdx.x) >> 5;
    int lane = threadIdx.x & 31;
    int mt = w >> 4;
    int nc = w & 15;

    v8f acc[4];
    const __bf16* Abase = pY + (size_t)mt * KT32 * 512 + lane * 16;
#pragma unroll
    for (int jn = 0; jn < 4; ++jn) {
        const __bf16* Bp = pWo + (size_t)(nc * 4 + jn) * KT32 * 512 + lane * 16;
        v8f c = {};
        v16bf a0 = *(const v16bf*)(Abase);
        v16bf b0 = *(const v16bf*)(Bp);
        for (int kt = 0; kt < KT32 - 1; ++kt) {
            v16bf a1 = *(const v16bf*)(Abase + (size_t)(kt + 1) * 512);
            v16bf b1 = *(const v16bf*)(Bp + (size_t)(kt + 1) * 512);
            c = wmma_bf16(a0, b0, c);
            a0 = a1; b0 = b1;
        }
        c = wmma_bf16(a0, b0, c);
        acc[jn] = c;
    }
    int half = lane >> 4, col = lane & 15;
#pragma unroll
    for (int i = 0; i < 8; ++i) {
        int m = mt * 16 + i + half * 8;
#pragma unroll
        for (int jn = 0; jn < 4; ++jn)
            out[(size_t)m * CH + nc * 64 + jn * 16 + col] = acc[jn][i];
    }
}

// ---------------------------------------------------------------------------
// Launch
// ---------------------------------------------------------------------------
extern "C" void kernel_launch(void* const* d_in, const int* in_sizes, int n_in,
                              void* d_out, int out_size, void* d_ws, size_t ws_size,
                              hipStream_t stream) {
    const float* x        = (const float*)d_in[0];
    const float* Wq       = (const float*)d_in[1];
    const float* Wk       = (const float*)d_in[2];
    const float* Wv       = (const float*)d_in[3];
    const float* Wo       = (const float*)d_in[4];
    const float* rope_cos = (const float*)d_in[5];
    const float* rope_sin = (const float*)d_in[6];
    float* out = (float*)d_out;

    const size_t MB = 1ull << 20;
    char* ws = (char*)d_ws;
    __bf16* pX   = (__bf16*)(ws + 0 * MB);     // 8 MB  (4096x1024 bf16 A-frags)
    __bf16* pWq  = (__bf16*)(ws + 8 * MB);     // 2 MB each
    __bf16* pWk  = (__bf16*)(ws + 10 * MB);
    __bf16* pWv  = (__bf16*)(ws + 12 * MB);
    __bf16* pWo  = (__bf16*)(ws + 14 * MB);
    __bf16* qbuf = (__bf16*)(ws + 16 * MB);    // 8 MB, (b,h,t,hd) bf16
    __bf16* kbuf = (__bf16*)(ws + 24 * MB);    // 8 MB, (b,h,t,hd) bf16
    __bf16* vTb  = (__bf16*)(ws + 32 * MB);    // 8 MB, (b,h,hd,t) bf16
    __bf16* pY   = (__bf16*)(ws + 40 * MB);    // 8 MB  (A-frags of attention out)
    // total: 48 MB of ws

    // 1) pack x into A-fragments: 8192 fragments, 4 waves/block
    pack_a_kernel<<<2048, 128, 0, stream>>>(x, pX, MROWS, CH);

    // 2) pack weights into B-fragments: 2048 fragments each
    pack_b_kernel<<<512, 128, 0, stream>>>(Wq, pWq, CH, CH);
    pack_b_kernel<<<512, 128, 0, stream>>>(Wk, pWk, CH, CH);
    pack_b_kernel<<<512, 128, 0, stream>>>(Wv, pWv, CH, CH);
    pack_b_kernel<<<512, 128, 0, stream>>>(Wo, pWo, CH, CH);

    // 3) fused QKV GEMM + RoPE + V transpose: 4096 waves, 8 waves/block
    qkv_kernel<<<512, 256, 0, stream>>>(pX, pWq, pWk, pWv, rope_cos, rope_sin,
                                        qbuf, kbuf, vTb);

    // 4) sliding-window flash attention: 4096 waves, 4 waves/block
    attn_kernel<<<1024, 128, 0, stream>>>(qbuf, kbuf, vTb, pY);

    // 5) output projection
    outproj_kernel<<<512, 256, 0, stream>>>(pY, pWo, out);
}
